// MultiHeadSelfAttention_26173530702435
// MI455X (gfx1250) — matrix operand order repaired and checked by reading, not compiled, not run on hardware
//
#include <hip/hip_runtime.h>
#include <hip/hip_bf16.h>

// ---------------------------------------------------------------------------
// MHSA for MI455X (gfx1250): bf16 WMMA projections + flash attention with
// async-to-LDS K/V staging. B=2, N=2048, DIM=1024, NH=16, head dim 64.
// ---------------------------------------------------------------------------

#define DIN   1024
#define SEQ   2048
#define NHEAD 16
#define HD    64
#define BSZ   2
#define ROWS  (BSZ * SEQ)        // 4096

typedef __attribute__((ext_vector_type(16))) __bf16 v16bf;
typedef __attribute__((ext_vector_type(8)))  __bf16 v8bf;
typedef __attribute__((ext_vector_type(4)))  __bf16 v4bf;
typedef __attribute__((ext_vector_type(8)))  float  v8f;
typedef __attribute__((ext_vector_type(4)))  float  v4f;

static __device__ __forceinline__ v16bf combine16(v8bf lo, v8bf hi) {
    return __builtin_shufflevector(lo, hi, 0,1,2,3,4,5,6,7,8,9,10,11,12,13,14,15);
}
static __device__ __forceinline__ v8f wmma_bf16(v16bf a, v16bf b, v8f c) {
    return __builtin_amdgcn_wmma_f32_16x16x32_bf16(false, a, false, b, (short)0, c,
                                                   false, false);
}
// async copy 16B: global (per-lane addr) -> LDS (per-lane offset). ASYNCcnt.
static __device__ __forceinline__ void async_g2l_b128(unsigned lds_off,
                                                      const __bf16* gptr) {
    asm volatile("global_load_async_to_lds_b128 %0, %1, off"
                 :: "v"(lds_off), "v"((unsigned long long)(uintptr_t)gptr)
                 : "memory");
}

// --------------------------- fp32 -> bf16 convert ---------------------------
__global__ __launch_bounds__(256)
void cvt_bf16_kernel(const float* __restrict__ in, __bf16* __restrict__ out, int n) {
    int i = (blockIdx.x * blockDim.x + threadIdx.x) * 4;
    if (i < n) {
        v4f v = *(const v4f*)(in + i);
        v4bf o = { (__bf16)v[0], (__bf16)v[1], (__bf16)v[2], (__bf16)v[3] };
        *(v4bf*)(out + i) = o;
    }
}

// ------------------- W[k][n] -> Wt[n][k] transpose + bf16 -------------------
__global__ __launch_bounds__(256)
void cvt_wt_kernel(const float* __restrict__ W, __bf16* __restrict__ Wt) {
    int n = blockIdx.x * 32 + threadIdx.x;   // coalesced read along n
    int k = blockIdx.y * 8 + threadIdx.y;
    Wt[(size_t)n * DIN + k] = (__bf16)W[(size_t)k * DIN + n];
}

// ------------------------------- QKV GEMM -----------------------------------
// O = x @ W + b.  xb: [4096,1024] bf16 row-major.  wt: [ncol,k] bf16 (W^T).
// Block tile 128x64, 4 waves, each wave 32 rows x 64 cols (8 accumulators).
// transposeV==0: out[bh][n][64] ; transposeV==1: out[bh][d][2048]
__global__ __launch_bounds__(128)
void qkv_proj_kernel(const __bf16* __restrict__ xb, const __bf16* __restrict__ wt,
                     const float* __restrict__ bias, __bf16* __restrict__ outp,
                     float scale, int transposeV) {
    const int mtile = blockIdx.x * 128;
    const int ntile = blockIdx.y * 64;
    const int wave  = threadIdx.x >> 5;
    const int lane  = threadIdx.x & 31;
    const int lmod  = lane & 15, lhi = lane >> 4;
    const int mrow  = mtile + wave * 32;

    v8f zero = {};
    v8f acc[2][4];
#pragma unroll
    for (int g = 0; g < 2; ++g)
#pragma unroll
        for (int t = 0; t < 4; ++t) acc[g][t] = zero;

    const __bf16* xrow0 = xb + (size_t)(mrow + lmod) * DIN + lhi * 8;
    const __bf16* xrow1 = xrow0 + (size_t)16 * DIN;

    for (int k = 0; k < DIN; k += 32) {
        v16bf af0 = combine16(*(const v8bf*)(xrow0 + k),
                              *(const v8bf*)(xrow0 + k + 16));
        v16bf af1 = combine16(*(const v8bf*)(xrow1 + k),
                              *(const v8bf*)(xrow1 + k + 16));
#pragma unroll
        for (int t = 0; t < 4; ++t) {
            const __bf16* wp = wt + (size_t)(ntile + t * 16 + lmod) * DIN + k + lhi * 8;
            v16bf bfrag = combine16(*(const v8bf*)wp, *(const v8bf*)(wp + 16));
            acc[0][t] = wmma_bf16(af0, bfrag, acc[0][t]);
            acc[1][t] = wmma_bf16(af1, bfrag, acc[1][t]);
        }
    }

#pragma unroll
    for (int g = 0; g < 2; ++g)
#pragma unroll
        for (int t = 0; t < 4; ++t) {
            const int col = ntile + t * 16 + lmod;
            const float bv = bias[col];
            const int h = col >> 6, d = col & 63;
#pragma unroll
            for (int r = 0; r < 8; ++r) {
                const int m = mrow + g * 16 + r + lhi * 8;   // row in [0,4096)
                const int b = m >> 11, n = m & (SEQ - 1);
                const __bf16 o = (__bf16)((acc[g][t][r] + bv) * scale);
                const size_t bh = (size_t)(b * NHEAD + h);
                if (!transposeV)
                    outp[(bh * SEQ + n) * HD + d] = o;
                else
                    outp[(bh * HD + d) * SEQ + n] = o;
            }
        }
}

// ---------------------------- flash attention -------------------------------
// grid = (SEQ/64, B*NH), block = 128 (4 waves). Wave w owns q rows
// [qtile + 16w, +16). K/V 64-key tiles staged once per block into LDS via
// global_load_async_to_lds_b128; all 4 waves read fragments from LDS.
__global__ __launch_bounds__(128)
void attn_kernel(const __bf16* __restrict__ Q, const __bf16* __restrict__ K,
                 const __bf16* __restrict__ Vt, float* __restrict__ out) {
    const int bh = blockIdx.y;
    const int b = bh >> 4, h = bh & 15;
    const int qtile = blockIdx.x * 64;
    const int wave = threadIdx.x >> 5;
    const int lane = threadIdx.x & 31;
    const int lmod = lane & 15, lhi = lane >> 4;

    __shared__ __bf16 kbuf[64][80];      // K tile: [key][d], stride 80
    __shared__ __bf16 vbuf[64][80];      // V^T tile: [d][key], stride 80
    __shared__ __bf16 pbuf[4][16][80];   // per-wave P tile (C->A transpose)

    const __bf16* Qh = Q  + (size_t)bh * SEQ * HD;
    const __bf16* Kh = K  + (size_t)bh * SEQ * HD;
    const __bf16* Vh = Vt + (size_t)bh * HD * SEQ;

    // Q A-fragments (16 rows x 64 d), resident in VGPRs for the whole loop.
    const __bf16* qp = Qh + (size_t)(qtile + wave * 16 + lmod) * HD + lhi * 8;
    v16bf qa[2];
#pragma unroll
    for (int f = 0; f < 2; ++f)
        qa[f] = combine16(*(const v8bf*)(qp + f * 32),
                          *(const v8bf*)(qp + f * 32 + 16));

    v8f zero = {};
    v8f accO[4];
#pragma unroll
    for (int t = 0; t < 4; ++t) accO[t] = zero;
    float mrun[8], lrun[8];
#pragma unroll
    for (int r = 0; r < 8; ++r) { mrun[r] = -1.0e30f; lrun[r] = 0.0f; }

    // per-thread staging assignment: 64B (4 x b128) of K and of V per tile
    const int srow  = threadIdx.x >> 1;            // 0..63
    const int shalf = (threadIdx.x & 1) * 32;      // column half (elements)
    const unsigned lds_k = (unsigned)(uintptr_t)&kbuf[srow][shalf];
    const unsigned lds_v = (unsigned)(uintptr_t)&vbuf[srow][shalf];

    for (int j = 0; j < SEQ; j += 64) {
        // ---- stage K/V tile: global -> LDS, ASYNCcnt-tracked ----
        const __bf16* gk = Kh + (size_t)(j + srow) * HD + shalf;
        const __bf16* gv = Vh + (size_t)srow * SEQ + j + shalf;
#pragma unroll
        for (int c = 0; c < 4; ++c) {
            async_g2l_b128(lds_k + c * 16, gk + c * 8);
            async_g2l_b128(lds_v + c * 16, gv + c * 8);
        }
        asm volatile("s_wait_asynccnt 0" ::: "memory");
        __syncthreads();

        // ---- S = Q . K^T from LDS ----
        v8f s[4];
#pragma unroll
        for (int t = 0; t < 4; ++t) {
            v8f sa = zero;
#pragma unroll
            for (int f = 0; f < 2; ++f) {
                const __bf16* kp = &kbuf[t * 16 + lmod][f * 32 + lhi * 8];
                v16bf kb = combine16(*(const v8bf*)kp, *(const v8bf*)(kp + 16));
                sa = wmma_bf16(qa[f], kb, sa);
            }
            s[t] = sa;
        }

        // ---- online softmax (row spans 16 lanes in C-layout) ----
        float mnew[8];
#pragma unroll
        for (int r = 0; r < 8; ++r) {
            float mx = fmaxf(fmaxf(s[0][r], s[1][r]), fmaxf(s[2][r], s[3][r]));
            mx = fmaxf(mx, __shfl_xor(mx, 1, 16));
            mx = fmaxf(mx, __shfl_xor(mx, 2, 16));
            mx = fmaxf(mx, __shfl_xor(mx, 4, 16));
            mx = fmaxf(mx, __shfl_xor(mx, 8, 16));
            mnew[r] = fmaxf(mrun[r], mx);
        }
        float rsum[8];
#pragma unroll
        for (int r = 0; r < 8; ++r) rsum[r] = 0.0f;
#pragma unroll
        for (int t = 0; t < 4; ++t)
#pragma unroll
            for (int r = 0; r < 8; ++r) {
                float p = __expf(s[t][r] - mnew[r]);
                s[t][r] = p;
                rsum[r] += p;
            }
#pragma unroll
        for (int r = 0; r < 8; ++r) {
            rsum[r] += __shfl_xor(rsum[r], 1, 16);
            rsum[r] += __shfl_xor(rsum[r], 2, 16);
            rsum[r] += __shfl_xor(rsum[r], 4, 16);
            rsum[r] += __shfl_xor(rsum[r], 8, 16);
            const float alpha = __expf(mrun[r] - mnew[r]);
            lrun[r] = lrun[r] * alpha + rsum[r];
            mrun[r] = mnew[r];
#pragma unroll
            for (int t = 0; t < 4; ++t) accO[t][r] *= alpha;
        }

        // ---- P: C-layout -> per-wave LDS tile -> A-layout fragments ----
        // DS ops from the same wave complete in order; an s_wait_dscnt is
        // enough to make the cross-lane transpose visible (no block barrier).
#pragma unroll
        for (int t = 0; t < 4; ++t)
#pragma unroll
            for (int r = 0; r < 8; ++r)
                pbuf[wave][r + lhi * 8][t * 16 + lmod] = (__bf16)s[t][r];
        asm volatile("s_wait_dscnt 0" ::: "memory");

        // ---- O += P . V from LDS ----
#pragma unroll
        for (int f = 0; f < 2; ++f) {
            const __bf16* pp = &pbuf[wave][lmod][f * 32 + lhi * 8];
            v16bf pa = combine16(*(const v8bf*)pp, *(const v8bf*)(pp + 16));
#pragma unroll
            for (int t = 0; t < 4; ++t) {
                const __bf16* vp = &vbuf[t * 16 + lmod][f * 32 + lhi * 8];
                v16bf vb = combine16(*(const v8bf*)vp, *(const v8bf*)(vp + 16));
                accO[t] = wmma_bf16(pa, vb, accO[t]);
            }
        }
        __syncthreads();   // protect kbuf/vbuf against next tile's async writes
    }

    // epilogue: normalize, write out[b][n][h*64+d] fp32
#pragma unroll
    for (int t = 0; t < 4; ++t)
#pragma unroll
        for (int r = 0; r < 8; ++r) {
            const int n = qtile + wave * 16 + r + lhi * 8;
            out[((size_t)b * SEQ + n) * (NHEAD * HD) + h * HD + t * 16 + lmod] =
                accO[t][r] / lrun[r];
        }
}

// ------------------------------- launcher -----------------------------------
extern "C" void kernel_launch(void* const* d_in, const int* in_sizes, int n_in,
                              void* d_out, int out_size, void* d_ws, size_t ws_size,
                              hipStream_t stream) {
    const float* x  = (const float*)d_in[0];
    const float* Wq = (const float*)d_in[1];
    const float* bq = (const float*)d_in[2];
    const float* Wk = (const float*)d_in[3];
    const float* bk = (const float*)d_in[4];
    const float* Wv = (const float*)d_in[5];
    const float* bv = (const float*)d_in[6];
    float* out = (float*)d_out;

    char* ws = (char*)d_ws;
    size_t off = 0;
    __bf16* xb  = (__bf16*)(ws + off); off += (size_t)ROWS * DIN * 2;     // 8 MiB
    __bf16* wtq = (__bf16*)(ws + off); off += (size_t)DIN * DIN * 2;      // 2 MiB
    __bf16* wtk = (__bf16*)(ws + off); off += (size_t)DIN * DIN * 2;
    __bf16* wtv = (__bf16*)(ws + off); off += (size_t)DIN * DIN * 2;
    __bf16* Qb  = (__bf16*)(ws + off); off += (size_t)ROWS * DIN * 2;     // 8 MiB
    __bf16* Kb  = (__bf16*)(ws + off); off += (size_t)ROWS * DIN * 2;
    __bf16* Vtb = (__bf16*)(ws + off); off += (size_t)ROWS * DIN * 2;
    if (off > ws_size) return;   // ~40 MiB scratch required

    const int nx = ROWS * DIN;
    cvt_bf16_kernel<<<nx / 4 / 256, 256, 0, stream>>>(x, xb, nx);
    dim3 wtg(DIN / 32, DIN / 8), wtb(32, 8);
    cvt_wt_kernel<<<wtg, wtb, 0, stream>>>(Wq, wtq);
    cvt_wt_kernel<<<wtg, wtb, 0, stream>>>(Wk, wtk);
    cvt_wt_kernel<<<wtg, wtb, 0, stream>>>(Wv, wtv);

    dim3 pg(ROWS / 128, DIN / 64);
    qkv_proj_kernel<<<pg, 128, 0, stream>>>(xb, wtq, bq, Qb, 0.125f, 0); // 1/sqrt(64)
    qkv_proj_kernel<<<pg, 128, 0, stream>>>(xb, wtk, bk, Kb, 1.0f, 0);
    qkv_proj_kernel<<<pg, 128, 0, stream>>>(xb, wtv, bv, Vtb, 1.0f, 1);

    dim3 ag(SEQ / 64, BSZ * NHEAD);
    attn_kernel<<<ag, 128, 0, stream>>>(Qb, Kb, Vtb, out);
}
